// GCNConvModel_52501680227004
// MI455X (gfx1250) — compile-verified, matrix operand order checked by reading, not run on hardware
//
#include <hip/hip_runtime.h>
#include <hip/hip_bf16.h>

#define N_NODES 100000
#define N_EDGES 320000
#define N_GRAPHS 512
#define F 256
#define F4 (F/4)

typedef __attribute__((ext_vector_type(16))) __bf16 v16bf;
typedef __attribute__((ext_vector_type(8)))  float  v8f;

__device__ __forceinline__ void atomAddF(float* p, float v) {
    __hip_atomic_fetch_add(p, v, __ATOMIC_RELAXED, __HIP_MEMORY_SCOPE_AGENT);
}

__device__ __forceinline__ unsigned pack_bf16(float a, float b) {
    union { __bf16 h[2]; unsigned u; } x;
    x.h[0] = (__bf16)a; x.h[1] = (__bf16)b;
    return x.u;
}

// ---------------- degree / dinv ----------------
__global__ __launch_bounds__(256) void init_deg_kernel(float* deg, int n) {
    int i = blockIdx.x * 256 + threadIdx.x;
    if (i < n) deg[i] = 1.0f;          // self loop
}

__global__ __launch_bounds__(256) void edge_deg_kernel(float* deg, const long long* dst, int ne) {
    int e = blockIdx.x * 256 + threadIdx.x;
    if (e < ne) atomAddF(&deg[dst[e]], 1.0f);
}

__global__ __launch_bounds__(256) void finish_dinv_kernel(float* dinv, int n) {
    int i = blockIdx.x * 256 + threadIdx.x;
    if (i < n) dinv[i] = rsqrtf(dinv[i]);   // deg >= 1 always
}

// ---------------- WMMA GEMM: Out[M,N] = (relu?)X[M,K] @ W[K,N] (+bias) ----------------
// block = 256 threads = 8 waves; block tile = 128(M) x N; K staged 32 at a time.
// LDS ping-pong + register-staged global loads overlap HBM with WMMA; B-fragments
// use a 2-deep register pipeline so DS latency hides under the matrix ops.
template <int NT, bool RELU>
__global__ __launch_bounds__(256) void gemm_wmma_kernel(
    const float* __restrict__ X, const float* __restrict__ W,
    const float* __restrict__ bias, float* __restrict__ Out, int M, int K)
{
    static_assert((NT & 1) == 0, "NT must be even");
    constexpr int N = NT * 16;
    __shared__ __bf16 ldsX[2][128 * 32];   // [m][k]  row stride 32 (64B)
    __shared__ __bf16 ldsW[2][N * 32];     // [n][k]  transposed, row stride 32

    const int tid  = threadIdx.x;
    const int wave = tid >> 5;
    const int lane = tid & 31;
    const int m0   = blockIdx.x * 128;

    // staging assignments
    const int xrow = tid >> 1;             // 0..127
    const int xkh  = (tid & 1) * 16;       // 0 or 16
    const int gm   = m0 + xrow;
    const int wn   = tid;                  // n column (active when < N)

    v8f acc[NT];
#pragma unroll
    for (int i = 0; i < NT; ++i)
#pragma unroll
        for (int j = 0; j < 8; ++j) acc[i][j] = 0.0f;

    float xr[16];
    float wr[32];

    // ---- issue global loads for slab 0 ----
    {
#pragma unroll
        for (int j = 0; j < 16; ++j) xr[j] = 0.0f;
        if (gm < M) {
            const float* xp = X + (size_t)gm * K + xkh;
#pragma unroll
            for (int j4 = 0; j4 < 4; ++j4) {
                float4 v = *(const float4*)(xp + j4 * 4);
                xr[j4 * 4 + 0] = v.x; xr[j4 * 4 + 1] = v.y;
                xr[j4 * 4 + 2] = v.z; xr[j4 * 4 + 3] = v.w;
            }
        }
        if (wn < N) {
#pragma unroll
            for (int k = 0; k < 32; ++k) wr[k] = W[(size_t)k * N + wn];
        }
    }

    int buf = 0;
    for (int kk = 0; kk < K; kk += 32, buf ^= 1) {
        // ---- convert + store staged registers into LDS buf ----
        {
            unsigned* lx = (unsigned*)&ldsX[buf][0];
            const int xbase = xrow * 16 + (xkh >> 1);     // dword index
            uint4 u0, u1;
            float v[16];
#pragma unroll
            for (int j = 0; j < 16; ++j) {
                float t = xr[j];
                if (RELU) t = t > 0.0f ? t : 0.0f;
                v[j] = t;
            }
            u0.x = pack_bf16(v[0], v[1]);  u0.y = pack_bf16(v[2], v[3]);
            u0.z = pack_bf16(v[4], v[5]);  u0.w = pack_bf16(v[6], v[7]);
            u1.x = pack_bf16(v[8], v[9]);  u1.y = pack_bf16(v[10], v[11]);
            u1.z = pack_bf16(v[12], v[13]); u1.w = pack_bf16(v[14], v[15]);
            *(uint4*)&lx[xbase + 0] = u0;
            *(uint4*)&lx[xbase + 4] = u1;

            if (wn < N) {
                unsigned* lw = (unsigned*)&ldsW[buf][0];
                const int wbase = wn * 16;                // dword index, [n][k]
#pragma unroll
                for (int q = 0; q < 4; ++q) {
                    uint4 u;
                    u.x = pack_bf16(wr[q * 8 + 0], wr[q * 8 + 1]);
                    u.y = pack_bf16(wr[q * 8 + 2], wr[q * 8 + 3]);
                    u.z = pack_bf16(wr[q * 8 + 4], wr[q * 8 + 5]);
                    u.w = pack_bf16(wr[q * 8 + 6], wr[q * 8 + 7]);
                    *(uint4*)&lw[wbase + q * 4] = u;
                }
            }
        }

        // ---- issue next slab's global loads (wait lands next iteration) ----
        const int kn = kk + 32;
        if (kn < K) {
#pragma unroll
            for (int j = 0; j < 16; ++j) xr[j] = 0.0f;
            if (gm < M) {
                const float* xp = X + (size_t)gm * K + kn + xkh;
#pragma unroll
                for (int j4 = 0; j4 < 4; ++j4) {
                    float4 v = *(const float4*)(xp + j4 * 4);
                    xr[j4 * 4 + 0] = v.x; xr[j4 * 4 + 1] = v.y;
                    xr[j4 * 4 + 2] = v.z; xr[j4 * 4 + 3] = v.w;
                }
            }
            if (wn < N) {
#pragma unroll
                for (int k = 0; k < 32; ++k) wr[k] = W[(size_t)(kn + k) * N + wn];
            }
        }

        __syncthreads();

        // ---- compute from LDS buf: 1 A-frag, NT B-frags, 2-deep B pipeline ----
        const int khalf = (lane >> 4) * 16;
        const v16bf a = *(const v16bf*)&ldsX[buf][(wave * 16 + (lane & 15)) * 32 + khalf];
        const __bf16* Wb = &ldsW[buf][(lane & 15) * 32 + khalf];

        v16bf b0 = *(const v16bf*)&Wb[0];
        v16bf b1 = *(const v16bf*)&Wb[16 * 32];
#pragma unroll
        for (int nt = 0; nt < NT; nt += 2) {
            acc[nt] = __builtin_amdgcn_wmma_f32_16x16x32_bf16(
                false, a, false, b0, (short)0, acc[nt], false, false);
            if (nt + 2 < NT) b0 = *(const v16bf*)&Wb[(size_t)(nt + 2) * 16 * 32];
            acc[nt + 1] = __builtin_amdgcn_wmma_f32_16x16x32_bf16(
                false, a, false, b1, (short)0, acc[nt + 1], false, false);
            if (nt + 3 < NT) b1 = *(const v16bf*)&Wb[(size_t)(nt + 3) * 16 * 32];
        }
    }

    // C/D layout: VGPR r holds M = r + 8*(lane>=16), N = lane%16
    const int nlane = lane & 15;
    const int mbase = m0 + wave * 16 + (lane >> 4) * 8;
#pragma unroll
    for (int nt = 0; nt < NT; ++nt) {
        const int n  = nt * 16 + nlane;
        const float bv = bias ? bias[n] : 0.0f;
#pragma unroll
        for (int r = 0; r < 8; ++r) {
            const int m = mbase + r;
            if (m < M) Out[(size_t)m * N + n] = acc[nt][r] + bv;
        }
    }
}

// ---------------- aggregation: out = D^-1/2 A D^-1/2 t + b ----------------
__global__ __launch_bounds__(256) void agg_init_kernel(
    const float4* __restrict__ t, float4* __restrict__ out,
    const float* __restrict__ dinv, const float* __restrict__ bias, int M)
{
    int gid = blockIdx.x * 256 + threadIdx.x;      // over M * F4
    if (gid >= M * F4) return;
    int i  = gid >> 6;
    int f4 = gid & 63;
    float di = dinv[i];
    float s  = di * di;
    float4 v = t[gid];
    const float4 bb = ((const float4*)bias)[f4];
    out[gid] = make_float4(v.x * s + bb.x, v.y * s + bb.y, v.z * s + bb.z, v.w * s + bb.w);
}

__global__ __launch_bounds__(256) void agg_edge_kernel(
    const float4* __restrict__ t, float* __restrict__ out,
    const long long* __restrict__ src, const long long* __restrict__ dst,
    const float* __restrict__ dinv, int ne)
{
    const int f4 = threadIdx.x & 63;
    const long long e = (long long)blockIdx.x * 4 + (threadIdx.x >> 6);
    if (e >= ne) return;
    const long long s = src[e], d = dst[e];
    const float nrm = dinv[s] * dinv[d];
    float4 v = t[s * F4 + f4];
    float* o = out + ((size_t)d * F + f4 * 4);
    atomAddF(o + 0, v.x * nrm);
    atomAddF(o + 1, v.y * nrm);
    atomAddF(o + 2, v.z * nrm);
    atomAddF(o + 3, v.w * nrm);
}

// ---------------- global mean pool (with fused ReLU on read) ----------------
__global__ __launch_bounds__(256) void pool_zero_kernel(float* sums, float* cnt) {
    int gid = blockIdx.x * 256 + threadIdx.x;
    if (gid < N_GRAPHS * F) sums[gid] = 0.0f;
    if (gid < N_GRAPHS)     cnt[gid]  = 0.0f;
}

__global__ __launch_bounds__(256) void pool_accum_kernel(
    const float4* __restrict__ h, const long long* __restrict__ batch,
    float* __restrict__ sums, float* __restrict__ cnt, int M)
{
    const int f4 = threadIdx.x & 63;
    const int i  = blockIdx.x * 4 + (threadIdx.x >> 6);
    if (i >= M) return;
    const long long g = batch[i];
    float4 v = h[(size_t)i * F4 + f4];
    v.x = v.x > 0.f ? v.x : 0.f;  v.y = v.y > 0.f ? v.y : 0.f;
    v.z = v.z > 0.f ? v.z : 0.f;  v.w = v.w > 0.f ? v.w : 0.f;
    float* o = sums + ((size_t)g * F + f4 * 4);
    atomAddF(o + 0, v.x); atomAddF(o + 1, v.y);
    atomAddF(o + 2, v.z); atomAddF(o + 3, v.w);
    if (f4 == 0) atomAddF(&cnt[g], 1.0f);
}

__global__ __launch_bounds__(256) void pool_div_kernel(
    const float* sums, const float* cnt, float* pooled)
{
    int gid = blockIdx.x * 256 + threadIdx.x;
    if (gid >= N_GRAPHS * F) return;
    float c = cnt[gid >> 8];
    pooled[gid] = sums[gid] / (c > 1.0f ? c : 1.0f);
}

// ---------------- launch ----------------
static inline size_t align256(size_t x) { return (x + 255) & ~(size_t)255; }

extern "C" void kernel_launch(void* const* d_in, const int* in_sizes, int n_in,
                              void* d_out, int out_size, void* d_ws, size_t ws_size,
                              hipStream_t stream)
{
    const float*      x     = (const float*)d_in[0];
    const long long*  eidx  = (const long long*)d_in[1];   // int64 [2, E]
    const long long*  batch = (const long long*)d_in[2];   // int64 [N]
    const float*      W1  = (const float*)d_in[3];
    const float*      b1  = (const float*)d_in[4];
    const float*      W2  = (const float*)d_in[5];
    const float*      b2  = (const float*)d_in[6];
    const float*      W3  = (const float*)d_in[7];
    const float*      b3  = (const float*)d_in[8];
    const float*      Wfc = (const float*)d_in[9];
    const float*      bfc = (const float*)d_in[10];
    const long long*  src = eidx;
    const long long*  dst = eidx + N_EDGES;

    char* ws = (char*)d_ws;
    size_t off = 0;
    const size_t hbytes = (size_t)N_NODES * F * sizeof(float);
    float* B0     = (float*)(ws + off); off += align256(hbytes);
    float* B1     = (float*)(ws + off); off += align256(hbytes);
    float* dinv   = (float*)(ws + off); off += align256((size_t)N_NODES * 4);
    float* sums   = (float*)(ws + off); off += align256((size_t)N_GRAPHS * F * 4);
    float* cnt    = (float*)(ws + off); off += align256((size_t)N_GRAPHS * 4);
    float* pooled = (float*)(ws + off); off += align256((size_t)N_GRAPHS * F * 4);

    const int nBlkNode = (N_NODES + 255) / 256;
    const int nBlkEdge = (N_EDGES + 255) / 256;
    const int gemmBlkM = (N_NODES + 127) / 128;
    const int aggBlk   = (N_NODES * F4 + 255) / 256;
    const int edgeBlk  = (N_EDGES + 3) / 4;
    const int nodeBlk4 = (N_NODES + 3) / 4;

    // degrees -> dinv (stored in dinv buffer)
    init_deg_kernel   <<<nBlkNode, 256, 0, stream>>>(dinv, N_NODES);
    edge_deg_kernel   <<<nBlkEdge, 256, 0, stream>>>(dinv, dst, N_EDGES);
    finish_dinv_kernel<<<nBlkNode, 256, 0, stream>>>(dinv, N_NODES);

    // layer 1: t = x @ W1 ; h1 = norm-agg(t) + b1   (ReLU deferred into layer-2 load)
    gemm_wmma_kernel<16, false><<<gemmBlkM, 256, 0, stream>>>(x,  W1, nullptr, B0, N_NODES, 128);
    agg_init_kernel <<<aggBlk,  256, 0, stream>>>((const float4*)B0, (float4*)B1, dinv, b1, N_NODES);
    agg_edge_kernel <<<edgeBlk, 256, 0, stream>>>((const float4*)B0, B1, src, dst, dinv, N_EDGES);

    // layer 2 (relu fused on GEMM load)
    gemm_wmma_kernel<16, true><<<gemmBlkM, 256, 0, stream>>>(B1, W2, nullptr, B0, N_NODES, 256);
    agg_init_kernel <<<aggBlk,  256, 0, stream>>>((const float4*)B0, (float4*)B1, dinv, b2, N_NODES);
    agg_edge_kernel <<<edgeBlk, 256, 0, stream>>>((const float4*)B0, B1, src, dst, dinv, N_EDGES);

    // layer 3
    gemm_wmma_kernel<16, true><<<gemmBlkM, 256, 0, stream>>>(B1, W3, nullptr, B0, N_NODES, 256);
    agg_init_kernel <<<aggBlk,  256, 0, stream>>>((const float4*)B0, (float4*)B1, dinv, b3, N_NODES);
    agg_edge_kernel <<<edgeBlk, 256, 0, stream>>>((const float4*)B0, B1, src, dst, dinv, N_EDGES);

    // mean pool (ReLU fused) + FC
    pool_zero_kernel <<<(N_GRAPHS * F + 255) / 256, 256, 0, stream>>>(sums, cnt);
    pool_accum_kernel<<<nodeBlk4, 256, 0, stream>>>((const float4*)B1, batch, sums, cnt, N_NODES);
    pool_div_kernel  <<<(N_GRAPHS * F + 255) / 256, 256, 0, stream>>>(sums, cnt, pooled);

    gemm_wmma_kernel<8, false><<<(N_GRAPHS + 127) / 128, 256, 0, stream>>>(
        pooled, Wfc, bfc, (float*)d_out, N_GRAPHS, 256);
}
